// FPNEncoder_73315091743288
// MI455X (gfx1250) — matrix operand order repaired and checked by reading, not compiled, not run on hardware
//
#include <hip/hip_runtime.h>
#include <hip/hip_bf16.h>

// ---------------- problem constants ----------------
#define NB 4        // batch
#define TT 120      // seq len
#define NN 1000     // nodes
#define CC 32       // conv channels
#define DD 40       // node embedding dim
#define KTOP 20     // top-k per row
#define PALPHA 0.05f
#define GALPHA 3.0f
#define MLP_COLS 64 // columns staged per TDM tile in mixprop MLP

typedef __attribute__((ext_vector_type(16))) _Float16 v16h;
typedef __attribute__((ext_vector_type(8)))  float    v8f;
typedef unsigned int       u32;
typedef unsigned long long u64;
typedef __attribute__((ext_vector_type(4))) u32 u32x4;
typedef __attribute__((ext_vector_type(8))) u32 u32x8;

// =====================================================================
// scale0: [B,1,N,120] conv k=120 -> [B,C,N,1]
// =====================================================================
__global__ void k_scale0(const float* __restrict__ x, const float* __restrict__ w,
                         const float* __restrict__ b, float* __restrict__ out) {
  int t = blockIdx.x * blockDim.x + threadIdx.x;
  if (t >= NB * CC * NN) return;
  int n = t % NN, c = (t / NN) % CC, bb = t / (NN * CC);
  float acc = b[c];
  for (int k = 0; k < TT; ++k) acc += x[(bb * TT + k) * NN + n] * w[c * TT + k];
  out[t] = acc;
}

// =====================================================================
// Cin=1 temporal conv (msc0): x[B,T,N] -> out[B,C,N,Lout]
// =====================================================================
__global__ void k_conv1(const float* __restrict__ x, const float* __restrict__ w,
                        const float* __restrict__ b, float* __restrict__ out,
                        int kw, int Lout) {
  long total = (long)NB * CC * NN * Lout;
  long t = (long)blockIdx.x * blockDim.x + threadIdx.x;
  if (t >= total) return;
  int l = (int)(t % Lout);
  long r = t / Lout;
  int n = (int)(r % NN), c = (int)((r / NN) % CC), bb = (int)(r / (NN * (long)CC));
  float acc = b[c];
  for (int k = 0; k < kw; ++k) acc += x[(bb * TT + l + k) * NN + n] * w[c * kw + k];
  out[t] = acc;
}

// =====================================================================
// Cin=32 temporal conv as implicit GEMM via WMMA f32<-f16 16x16x32.
// =====================================================================
__global__ void k_conv_wmma(const float* __restrict__ in, const float* __restrict__ w,
                            const float* __restrict__ bias, float* __restrict__ out,
                            int kw, int Lin, int Lout) {
  const int NLo = NN * Lout;
  const int colTiles = (NLo + 15) / 16;
  int wid = (blockIdx.x * blockDim.x + threadIdx.x) >> 5;
  int total = colTiles * 2 * NB;
  if (wid >= total) return;                       // wave-uniform
  int ct = wid % colTiles, mt = (wid / colTiles) & 1, bb = wid / (colTiles * 2);
  int lane = threadIdx.x & 31, lo = lane & 15, hi = lane >> 4;
  int col = ct * 16 + lo;
  bool cok = col < NLo;
  int vn = cok ? col / Lout : 0, vl = cok ? col % Lout : 0;
  int mrow = mt * 16 + lo;                        // A-operand row (M)
  const float* inb = in + (long)bb * CC * NN * Lin;
  v8f acc = {};
  for (int kk = 0; kk < kw; ++kk) {
    v16h A, Bv;
#pragma unroll
    for (int p = 0; p < 8; ++p) {
      int k0 = ((p >> 2) << 4) + (hi << 3) + ((p & 3) << 1);   // ci
      A[2 * p]     = (_Float16)w[(mrow * CC + k0) * kw + kk];
      A[2 * p + 1] = (_Float16)w[(mrow * CC + k0 + 1) * kw + kk];
    }
#pragma unroll
    for (int j = 0; j < 16; ++j) {
      int ci = ((j >> 3) << 4) + (hi << 3) + (j & 7);
      Bv[j] = (_Float16)(cok ? inb[((long)ci * NN + vn) * Lin + (vl + kk)] : 0.f);
    }
    acc = __builtin_amdgcn_wmma_f32_16x16x32_f16(false, A, false, Bv, (short)0, acc,
                                                 false, false);
  }
  float* outb = out + (long)bb * CC * NLo;
#pragma unroll
  for (int r = 0; r < 8; ++r) {
    int m = mt * 16 + r + (hi << 3);
    if (cok) outb[(long)m * NLo + col] = acc[r] + bias[m];
  }
}

// =====================================================================
// graph linear: nout = tanh(ALPHA * ((nin*s) @ W^T + b)), [N,D]x[D,D]
// =====================================================================
__global__ void k_glin(const float* __restrict__ nin, const float* __restrict__ W,
                       const float* __restrict__ b, float s, float* __restrict__ nout) {
  int t = blockIdx.x * blockDim.x + threadIdx.x;
  if (t >= NN * DD) return;
  int o = t % DD, r = t / DD;
  float acc = b[o];
  for (int k = 0; k < DD; ++k) acc += nin[r * DD + k] * s * W[o * DD + k];
  nout[t] = tanhf(GALPHA * acc);
}

// =====================================================================
// G = n1 @ n2^T  (WMMA, K=40 zero-padded to 64)
// =====================================================================
__global__ void k_gemm_nt(const float* __restrict__ n1, const float* __restrict__ n2,
                          float* __restrict__ G) {
  const int tiles = (NN + 15) / 16;               // 63
  int wid = (blockIdx.x * blockDim.x + threadIdx.x) >> 5;
  if (wid >= tiles * tiles) return;
  int ct = wid % tiles, mt = wid / tiles;
  int lane = threadIdx.x & 31, lo = lane & 15, hi = lane >> 4;
  int m = mt * 16 + lo, col = ct * 16 + lo;
  bool mok = m < NN, cok = col < NN;
  v8f acc = {};
#pragma unroll
  for (int kb = 0; kb < 2; ++kb) {
    v16h A, Bv;
#pragma unroll
    for (int p = 0; p < 8; ++p) {
      int k0 = kb * 32 + ((p >> 2) << 4) + (hi << 3) + ((p & 3) << 1);
      A[2 * p]     = (_Float16)((mok && k0 < DD)     ? n1[m * DD + k0]     : 0.f);
      A[2 * p + 1] = (_Float16)((mok && k0 + 1 < DD) ? n1[m * DD + k0 + 1] : 0.f);
    }
#pragma unroll
    for (int j = 0; j < 16; ++j) {
      int k = kb * 32 + ((j >> 3) << 4) + (hi << 3) + (j & 7);
      Bv[j] = (_Float16)((cok && k < DD) ? n2[col * DD + k] : 0.f);
    }
    acc = __builtin_amdgcn_wmma_f32_16x16x32_f16(false, A, false, Bv, (short)0, acc,
                                                 false, false);
  }
#pragma unroll
  for (int r = 0; r < 8; ++r) {
    int mm = mt * 16 + r + (hi << 3);
    if (mm < NN && cok) G[(long)mm * NN + col] = acc[r];
  }
}

// adjacency value recomputed on the fly from G
__device__ __forceinline__ float adjval(const float* __restrict__ G, int v, int w) {
  float a = G[(long)v * NN + w] - G[(long)w * NN + v];
  float t = tanhf(GALPHA * a);
  return t > 0.f ? t : 0.f;
}

// per-row K-th-largest threshold via binary search; one wave per row
__global__ void k_thresh(const float* __restrict__ G, float* __restrict__ thr) {
  int v = blockIdx.x;
  int lane = threadIdx.x;
  float lo = 0.f, hi = 1.f;
  for (int it = 0; it < 24; ++it) {
    float mid = 0.5f * (lo + hi);
    int cnt = 0;
    for (int w = lane; w < NN; w += 32)
      if (adjval(G, v, w) > mid) cnt++;
    for (int off = 16; off; off >>= 1) cnt += __shfl_xor(cnt, off, 32);
    if (cnt >= KTOP) lo = mid; else hi = mid;     // uniform across wave
  }
  if (lane == 0) thr[v] = lo;
}

__global__ void k_zero_i(int* p, int n) {
  int t = blockIdx.x * blockDim.x + threadIdx.x;
  if (t < n) p[t] = 0;
}

// extract forward sparse rows of a=(mask(adj)+I) row-normalized
__global__ void k_extract(const float* __restrict__ G, const float* __restrict__ thr,
                          int* __restrict__ fidx, float* __restrict__ fraw,
                          float* __restrict__ fval, int* __restrict__ tcnt) {
  int v = blockIdx.x * blockDim.x + threadIdx.x;
  if (v >= NN) return;
  float th = thr[v];
  int cnt = 1;
  fidx[v * 21] = v; fraw[v * 21] = 0.f;
  float rowsum = 1.f;                              // +I diagonal
  for (int w = 0; w < NN && cnt < 21; ++w) {
    float m = adjval(G, v, w);
    if (m > th) {
      fidx[v * 21 + cnt] = w;
      fraw[v * 21 + cnt] = m;
      rowsum += m;
      cnt++;
      atomicAdd(&tcnt[w], 1);
    }
  }
  for (int j = cnt; j < 21; ++j) { fidx[v * 21 + j] = v; fraw[v * 21 + j] = 0.f; }
  float inv = 1.f / rowsum;
  fval[v * 21] = inv;
  for (int j = 1; j < 21; ++j) fval[v * 21 + j] = fraw[v * 21 + j] * inv;
}

__global__ void k_scan(const int* __restrict__ tcnt, int* __restrict__ tptr,
                       int* __restrict__ tcur) {
  if (threadIdx.x == 0 && blockIdx.x == 0) {
    int s = 0;
    for (int w = 0; w < NN; ++w) { tptr[w] = s; tcur[w] = s; s += tcnt[w]; }
    tptr[NN] = s;
  }
}

__global__ void k_fill(const int* __restrict__ fidx, const float* __restrict__ fraw,
                       int* __restrict__ tcur, int* __restrict__ tidx,
                       float* __restrict__ tval) {
  int v = blockIdx.x * blockDim.x + threadIdx.x;
  if (v >= NN) return;
  for (int j = 1; j < 21; ++j) {
    float m = fraw[v * 21 + j];
    if (m > 0.f) {
      int w = fidx[v * 21 + j];
      int pos = atomicAdd(&tcur[w], 1);
      tidx[pos] = v; tval[pos] = m;
    }
  }
}

__global__ void k_tinv(const int* __restrict__ tptr, const float* __restrict__ tval,
                       float* __restrict__ tinv) {
  int v = blockIdx.x * blockDim.x + threadIdx.x;
  if (v >= NN) return;
  float s = 1.f;
  for (int e = tptr[v]; e < tptr[v + 1]; ++e) s += tval[e];
  tinv[v] = 1.f / s;
}

// sparse mix-hop propagation, forward
__global__ void k_prop_fwd(const float* __restrict__ x, const float* __restrict__ h,
                           const int* __restrict__ fidx, const float* __restrict__ fval,
                           float* __restrict__ hn, int L) {
  long total = (long)NB * CC * NN * L;
  long t = (long)blockIdx.x * blockDim.x + threadIdx.x;
  if (t >= total) return;
  int l = (int)(t % L);
  long r = t / L;
  int v = (int)(r % NN);
  long bc = r / NN;
  const float* hb = h + bc * NN * (long)L;
  float acc = 0.f;
#pragma unroll 3
  for (int j = 0; j < 21; ++j)
    acc += fval[v * 21 + j] * hb[(long)fidx[v * 21 + j] * L + l];
  hn[t] = PALPHA * x[t] + (1.f - PALPHA) * acc;
}

// transpose direction via CSC
__global__ void k_prop_t(const float* __restrict__ x, const float* __restrict__ h,
                         const int* __restrict__ tptr, const int* __restrict__ tidx,
                         const float* __restrict__ tval, const float* __restrict__ tinv,
                         float* __restrict__ hn, int L) {
  long total = (long)NB * CC * NN * L;
  long t = (long)blockIdx.x * blockDim.x + threadIdx.x;
  if (t >= total) return;
  int l = (int)(t % L);
  long r = t / L;
  int v = (int)(r % NN);
  long bc = r / NN;
  const float* hb = h + bc * NN * (long)L;
  float acc = hb[(long)v * L + l];
  int e0 = tptr[v], e1 = tptr[v + 1];
  for (int e = e0; e < e1; ++e) acc += tval[e] * hb[(long)tidx[e] * L + l];
  hn[t] = PALPHA * x[t] + (1.f - PALPHA) * tinv[v] * acc;
}

// =====================================================================
// mixprop 1x1 MLP, TDM-staged WMMA GEMM.
// Workgroup (4 waves) computes a 32(M) x 64(N) output block.
// Wave 0 issues three tensor_load_to_lds (one per hop buffer) pulling a
// 32x64 fp32 tile each into LDS (TENSORcnt-tracked), waits, barrier;
// all waves then feed WMMA B-operands from LDS. Edge tiles are
// hardware-zero-filled via tensor_dim0 OOB semantics.
// =====================================================================
__global__ void __launch_bounds__(128)
k_mlp3_tdm(const float* __restrict__ h0, const float* __restrict__ h1,
           const float* __restrict__ h2, const float* __restrict__ w,
           const float* __restrict__ bias, float* __restrict__ out,
           int L, int accflag) {
  __shared__ float sh[3 * CC * MLP_COLS];          // 24 KB
  const int NL = NN * L;
  const int colBlocks = (NL + MLP_COLS - 1) / MLP_COLS;
  int cb = blockIdx.x % colBlocks;
  int bb = blockIdx.x / colBlocks;
  int colbase = cb * MLP_COLS;
  const float* Hs[3] = { h0 + (long)bb * CC * NL, h1 + (long)bb * CC * NL,
                         h2 + (long)bb * CC * NL };
  int wave = threadIdx.x >> 5;
  int lane = threadIdx.x & 31, lo = lane & 15, hi = lane >> 4;

  if (wave == 0) {                                 // wave-uniform branch
    u32 ldsbase = (u32)(size_t)(&sh[0]);
    u32 remain = (u32)(NL - colbase);              // hardware OOB -> zeros
#pragma unroll
    for (int jb = 0; jb < 3; ++jb) {
      u64 gaddr = (u64)(size_t)Hs[jb] + (u64)colbase * 4u;
      u32 lds_addr = ldsbase + (u32)(jb * CC * MLP_COLS * 4);
      // ---- D# group0: count=1, lds_addr, global_addr, type=2 ----
      u64 q0 = 1ull | ((u64)lds_addr << 32);
      u64 q1 = (gaddr & 0x01FFFFFFFFFFFFFFull) | (2ull << 62);
      u32x4 g0 = { (u32)q0, (u32)(q0 >> 32), (u32)q1, (u32)(q1 >> 32) };
      // ---- D# group1: data_size=4B, 2D tile 64x32, stride NL ----
      u64 td0 = remain;                            // tensor_dim0 (elems)
      u64 td1 = CC;                                // tensor_dim1
      u64 t0  = MLP_COLS;                          // tile_dim0
      u64 t1  = CC;                                // tile_dim1
      u64 s0  = (u64)NL;                           // tensor_dim0_stride
      u64 p0 = (0x2ull << 16) | ((td0 & 0xffffull) << 48);
      u64 p1 = (td0 >> 16) | (td1 << 16) | (t0 << 48);
      u64 p2 = t1 | ((s0 & 0xffffffffull) << 32);
      u64 p3 = (s0 >> 32);
      u32x8 g1 = { (u32)p0, (u32)(p0 >> 32), (u32)p1, (u32)(p1 >> 32),
                   (u32)p2, (u32)(p2 >> 32), (u32)p3, (u32)(p3 >> 32) };
      asm volatile("tensor_load_to_lds %0, %1" :: "s"(g0), "s"(g1) : "memory");
    }
    __builtin_amdgcn_s_wait_tensorcnt(0);          // TDM tiles landed in LDS
  }
  __syncthreads();

  int collocal = wave * 16 + lo;
  int col = colbase + collocal;
  bool cok = col < NL;
  v8f acc0 = {}, acc1 = {};
#pragma unroll
  for (int jb = 0; jb < 3; ++jb) {
    v16h A0, A1, Bv;
#pragma unroll
    for (int p = 0; p < 8; ++p) {
      int k0 = ((p >> 2) << 4) + (hi << 3) + ((p & 3) << 1);
      A0[2 * p]     = (_Float16)w[lo * (3 * CC) + jb * CC + k0];
      A0[2 * p + 1] = (_Float16)w[lo * (3 * CC) + jb * CC + k0 + 1];
      A1[2 * p]     = (_Float16)w[(16 + lo) * (3 * CC) + jb * CC + k0];
      A1[2 * p + 1] = (_Float16)w[(16 + lo) * (3 * CC) + jb * CC + k0 + 1];
    }
#pragma unroll
    for (int j = 0; j < 16; ++j) {
      int ci = ((j >> 3) << 4) + (hi << 3) + (j & 7);
      Bv[j] = (_Float16)sh[jb * CC * MLP_COLS + ci * MLP_COLS + collocal];
    }
    acc0 = __builtin_amdgcn_wmma_f32_16x16x32_f16(false, A0, false, Bv, (short)0, acc0,
                                                  false, false);
    acc1 = __builtin_amdgcn_wmma_f32_16x16x32_f16(false, A1, false, Bv, (short)0, acc1,
                                                  false, false);
  }
  float* ob = out + (long)bb * CC * NL;
#pragma unroll
  for (int r = 0; r < 8; ++r) {
    int m0 = r + (hi << 3);
    int m1 = 16 + m0;
    if (cok) {
      long o0 = (long)m0 * NL + col;
      long o1 = (long)m1 * NL + col;
      float v0 = acc0[r] + bias[m0];
      float v1 = acc1[r] + bias[m1];
      ob[o0] = accflag ? (ob[o0] + v0) : v0;
      ob[o1] = accflag ? (ob[o1] + v1) : v1;
    }
  }
}

// =====================================================================
// host orchestration
// =====================================================================
extern "C" void kernel_launch(void* const* d_in, const int* in_sizes, int n_in,
                              void* d_out, int out_size, void* d_ws, size_t ws_size,
                              hipStream_t stream) {
  const float* x    = (const float*)d_in[0];
  const float* s0w  = (const float*)d_in[1];
  const float* s0b  = (const float*)d_in[2];
  const float* mscw[4] = { (const float*)d_in[3], (const float*)d_in[5],
                           (const float*)d_in[7], (const float*)d_in[9] };
  const float* mscb[4] = { (const float*)d_in[4], (const float*)d_in[6],
                           (const float*)d_in[8], (const float*)d_in[10] };
  const float* emb1  = (const float*)d_in[11];
  const float* emb2  = (const float*)d_in[12];
  const float* lin1w = (const float*)d_in[13];
  const float* lin1b = (const float*)d_in[14];
  const float* lin2w = (const float*)d_in[15];
  const float* lin2b = (const float*)d_in[16];
  const float* mp1w  = (const float*)d_in[17];
  const float* mp1b  = (const float*)d_in[18];
  const float* mp2w  = (const float*)d_in[19];
  const float* mp2b  = (const float*)d_in[20];

  const int KSET[4] = { 14, 7, 6, 3 };
  const float SSET[4] = { 1.0f, 0.8f, 0.6f, 0.5f };
  const int Ls[4] = { 107, 101, 96, 94 };
  const long SCALE_ELEMS = (long)NB * CC * NN * 107;

  // ---- workspace carve ----
  float* F = (float*)d_ws;
  size_t off = 0;
  auto alloc = [&](size_t n) { float* p = F + off; off += (n + 63) & ~(size_t)63; return p; };
  float* sA   = alloc(SCALE_ELEMS);
  float* sB   = alloc(SCALE_ELEMS);
  float* h1   = alloc(SCALE_ELEMS);
  float* h2   = alloc(SCALE_ELEMS);
  float* G1   = alloc((size_t)NN * NN);
  float* n1a  = alloc(NN * DD);
  float* n1b  = alloc(NN * DD);
  float* n2a  = alloc(NN * DD);
  float* n2b  = alloc(NN * DD);
  float* thr  = alloc(NN);
  float* fraw = alloc(NN * 21);
  float* fval = alloc(NN * 21);
  float* tval = alloc(NN * KTOP);
  float* tinv = alloc(NN);
  int* I = (int*)(F + off);
  int* fidx = I;            I += (NN * 21 + 63) & ~63;
  int* tcnt = I;            I += (NN + 63) & ~63;
  int* tptr = I;            I += (NN + 1 + 63) & ~63;
  int* tcur = I;            I += (NN + 63) & ~63;
  int* tidx = I;

  // ---- scale0 -> d_out ----
  k_scale0<<<(NB * CC * NN + 255) / 256, 256, 0, stream>>>(x, s0w, s0b, (float*)d_out);

  // ---- msc0 (Cin=1, k=14): x -> sA ----
  {
    long tot = (long)NB * CC * NN * Ls[0];
    k_conv1<<<(int)((tot + 255) / 256), 256, 0, stream>>>(x, mscw[0], mscb[0], sA,
                                                          KSET[0], Ls[0]);
  }

  float* cur = sA;
  float* nxt = sB;
  const float* n1p = emb1;
  const float* n2p = emb2;
  float* n1buf[2] = { n1a, n1b };
  float* n2buf[2] = { n2a, n2b };
  float* outp = (float*)d_out + (long)NB * CC * NN;

  for (int i = 0; i < 4; ++i) {
    int L = Ls[i];
    long tot = (long)NB * CC * NN * L;
    int eb = (int)((tot + 255) / 256);

    // ---- graph construction for scale i ----
    k_glin<<<(NN * DD + 255) / 256, 256, 0, stream>>>(n1p, lin1w + i * DD * DD,
                                                      lin1b + i * DD, SSET[i], n1buf[i & 1]);
    k_glin<<<(NN * DD + 255) / 256, 256, 0, stream>>>(n2p, lin2w + i * DD * DD,
                                                      lin2b + i * DD, SSET[i], n2buf[i & 1]);
    n1p = n1buf[i & 1]; n2p = n2buf[i & 1];

    const int gt = (NN + 15) / 16;
    int gwaves = gt * gt;
    k_gemm_nt<<<(gwaves + 3) / 4, 128, 0, stream>>>(n1p, n2p, G1);

    k_thresh<<<NN, 32, 0, stream>>>(G1, thr);
    k_zero_i<<<(NN + 255) / 256, 256, 0, stream>>>(tcnt, NN);
    k_extract<<<(NN + 127) / 128, 128, 0, stream>>>(G1, thr, fidx, fraw, fval, tcnt);
    k_scan<<<1, 32, 0, stream>>>(tcnt, tptr, tcur);
    k_fill<<<(NN + 127) / 128, 128, 0, stream>>>(fidx, fraw, tcur, tidx, tval);
    k_tinv<<<(NN + 127) / 128, 128, 0, stream>>>(tptr, tval, tinv);

    int NL = NN * L;
    int colBlocks = (NL + MLP_COLS - 1) / MLP_COLS;

    // ---- mixprop with adj (forward, sparse rows) ----
    k_prop_fwd<<<eb, 256, 0, stream>>>(cur, cur, fidx, fval, h1, L);
    k_prop_fwd<<<eb, 256, 0, stream>>>(cur, h1, fidx, fval, h2, L);
    k_mlp3_tdm<<<colBlocks * NB, 128, 0, stream>>>(cur, h1, h2, mp1w + i * CC * 3 * CC,
                                                   mp1b + i * CC, outp, L, 0);

    // ---- mixprop with adj^T (CSC) ----
    k_prop_t<<<eb, 256, 0, stream>>>(cur, cur, tptr, tidx, tval, tinv, h1, L);
    k_prop_t<<<eb, 256, 0, stream>>>(cur, h1, tptr, tidx, tval, tinv, h2, L);
    k_mlp3_tdm<<<colBlocks * NB, 128, 0, stream>>>(cur, h1, h2, mp2w + i * CC * 3 * CC,
                                                   mp2b + i * CC, outp, L, 1);

    outp += tot;

    // ---- next pyramid level conv (WMMA implicit GEMM) ----
    if (i < 3) {
      int Lout = Ls[i + 1], kw = KSET[i + 1];
      int NLo = NN * Lout, cT2 = (NLo + 15) / 16, cwaves = cT2 * 2 * NB;
      k_conv_wmma<<<(cwaves + 3) / 4, 128, 0, stream>>>(cur, mscw[i + 1], mscb[i + 1],
                                                        nxt, kw, L, Lout);
      float* tmp = cur; cur = nxt; nxt = tmp;
    }
  }
}
